// Individual_74749610819588
// MI455X (gfx1250) — matrix operand order repaired and checked by reading, not compiled, not run on hardware
//
#include <hip/hip_runtime.h>
#include <hip/hip_bf16.h>

#define INPUT_DIM   512
#define NUM_MID     512
#define NUM_NODES   1024
#define OUT_DIM     64
#define BATCH       32768
#define TILE_M      64          // batch rows per workgroup
#define BLOCK_THREADS 128       // 4 waves (wave32)

typedef __attribute__((ext_vector_type(16))) __bf16    v16bf;
typedef __attribute__((ext_vector_type(8)))  float     v8f;
typedef __attribute__((ext_vector_type(8)))  short     short8;
typedef __attribute__((ext_vector_type(4)))  unsigned  uint4v;
typedef __attribute__((ext_vector_type(8)))  int       int8v;
typedef __attribute__((ext_vector_type(4)))  int       int4v;

union Frag { short8 h[2]; v16bf v; };

__device__ __forceinline__ unsigned short f2bf(float f) {
    unsigned u = __float_as_uint(f);
    unsigned r = u + 0x7FFFu + ((u >> 16) & 1u);   // round-to-nearest-even
    return (unsigned short)(r >> 16);
}
__device__ __forceinline__ float bf2f(unsigned short h) {
    return __uint_as_float(((unsigned)h) << 16);
}
__device__ __forceinline__ float sigm(float t) {
    return __builtin_amdgcn_rcpf(1.0f + __expf(-t));   // v_rcp_f32 + v_exp_f32
}

// ---------------------------------------------------------------------------
// Prep: effT[n][k] = bf16(weight[k][n] * connectivity[k][n]),  n<512, k<1024
// Transposed so a 16-column weight slab is one contiguous 32 KB region.
// ---------------------------------------------------------------------------
__global__ void build_effT(const float* __restrict__ w,
                           const int*   __restrict__ conn,
                           unsigned short* __restrict__ effT) {
    int idx = blockIdx.x * blockDim.x + threadIdx.x;
    if (idx >= NUM_MID * NUM_NODES) return;
    int n = idx >> 10;        // weight column (mid node)  [0,512)
    int k = idx & 1023;       // weight row (source node)  [0,1024)
    float v = w[(size_t)k * NUM_MID + n] * (float)conn[(size_t)k * NUM_MID + n];
    effT[idx] = f2bf(v);
}

// ---------------------------------------------------------------------------
// Main: per-WG batch tile of 64 rows runs the full 512-step recurrence.
// Weight slabs are DMA'd to LDS by the Tensor Data Mover (TENSORcnt),
// dense part via v_wmma_f32_16x16x32_bf16, 16-wide triangular tail via
// ds_bpermute broadcasts + VALU sigmoid/FMA.
// ---------------------------------------------------------------------------
__global__ void __launch_bounds__(BLOCK_THREADS)
recurrent_kernel(const float* __restrict__ x,
                 const unsigned short* __restrict__ effT,
                 const float* __restrict__ bias,
                 const int*   __restrict__ exist,
                 float* __restrict__ out) {
    __shared__ __align__(16) unsigned short sig[TILE_M * NUM_NODES]; // 128 KB bf16
    __shared__ __align__(16) unsigned short slab[16 * NUM_NODES];    //  32 KB bf16

    const int tid  = threadIdx.x;
    const int lane = tid & 31;
    const int wave = tid >> 5;        // 0..3 -> 16-row tile
    const int half = lane >> 4;       // 0/1
    const int l16  = lane & 15;
    const int row0 = blockIdx.x * TILE_M;

    // ---- stage x tile as bf16; zero mid region ----
    for (int idx = tid; idx < TILE_M * (INPUT_DIM / 4); idx += BLOCK_THREADS) {
        int r = idx >> 7, c4 = idx & 127;
        const float4 v = ((const float4*)(x + (size_t)(row0 + r) * INPUT_DIM))[c4];
        unsigned lo = (unsigned)f2bf(v.x) | ((unsigned)f2bf(v.y) << 16);
        unsigned hi = (unsigned)f2bf(v.z) | ((unsigned)f2bf(v.w) << 16);
        unsigned* p = (unsigned*)&sig[r * NUM_NODES + c4 * 4];
        p[0] = lo; p[1] = hi;
    }
    unsigned* sig32 = (unsigned*)sig;
    for (int idx = tid; idx < TILE_M * (NUM_MID / 2); idx += BLOCK_THREADS) {
        int r = idx >> 8, c = idx & 255;
        sig32[r * (NUM_NODES / 2) + (INPUT_DIM / 2) + c] = 0u;
    }

    // LDS byte offset of the slab (generic shared addr low 32 bits == LDS offset)
    const unsigned slab_lds = (unsigned)(unsigned long long)(&slab[0]);

    // ---- 32 sequential blocks of 16 output columns ----
    for (int cb = 0; cb < NUM_MID / 16; ++cb) {
        const int cbase = cb * 16;

        // ---- TDM: DMA the 16-column weight slab (32 KB contiguous) to LDS ----
        if (wave == 0) {
            unsigned long long gaddr =
                (unsigned long long)(effT + (size_t)cbase * NUM_NODES);
            uint4v g0;
            g0[0] = 1u;                                    // count=1, user D#
            g0[1] = slab_lds;                              // lds_addr (bytes)
            g0[2] = (unsigned)(gaddr & 0xFFFFFFFFu);       // global_addr[31:0]
            g0[3] = (unsigned)((gaddr >> 32) & 0x1FFFFFFu) // global_addr[56:32]
                    | (2u << 30);                          // type = 2 (image)
            int8v g1;
            g1[0] = 0x00010000;          // wg_mask=0, data_size=2B
            g1[1] = 0;                   // no atomic barrier
            g1[2] = 0x00010008;          // tensor_dim0=524288, tensor_dim1=1
            g1[3] = 0x40000000;          // tile_dim0 = 16384 elements (32 KB)
            g1[4] = 1;                   // tile_dim1 = 1
            g1[5] = 0x00080000;          // tensor_dim0_stride = 524288
            g1[6] = 0;
            g1[7] = 0;
            int4v gz = {0, 0, 0, 0};
#if defined(__clang_major__) && __clang_major__ >= 23
            int8v gz8 = {0, 0, 0, 0, 0, 0, 0, 0};
            __builtin_amdgcn_tensor_load_to_lds(g0, g1, gz, gz, gz8, 0);
#else
            __builtin_amdgcn_tensor_load_to_lds(g0, g1, gz, gz, 0);
#endif
            __builtin_amdgcn_s_wait_tensorcnt(0);
        }
        __syncthreads();   // slab (and, for cb==0, the x staging) visible to all

        if (cb + 1 < NUM_MID / 16)
            __builtin_prefetch(effT + (size_t)(cbase + 16) * NUM_NODES, 0, 3);

        // acc starts at bias (per column = lane%16)
        v8f acc;
        const float bcol = bias[cbase + l16];
        #pragma unroll
        for (int v = 0; v < 8; ++v) acc[v] = bcol;

        // dense GEMM over all already-valid signal columns (round K up to 32;
        // over-read columns are still zero in LDS, so they contribute nothing)
        const int ksteps = (INPUT_DIM + cbase + 31) >> 5;
        for (int ks = 0; ks < ksteps; ++ks) {
            const int kbase = ks * 32;
            Frag A, B;
            // A (16x32 bf16): lanes<16 hold K {0-7,16-23}, lanes>=16 {8-15,24-31}
            const unsigned short* ap =
                &sig[(wave * 16 + l16) * NUM_NODES + kbase + half * 8];
            A.h[0] = *(const short8*)ap;
            A.h[1] = *(const short8*)(ap + 16);
            // B (32x16 bf16): lanes<16 hold K {0-7,16-23}, lanes>=16 {8-15,24-31}
            const unsigned short* bp = &slab[l16 * NUM_NODES + kbase + half * 8];
            B.h[0] = *(const short8*)bp;
            B.h[1] = *(const short8*)(bp + 16);
            acc = __builtin_amdgcn_wmma_f32_16x16x32_bf16(
                false, A.v, false, B.v, (short)0, acc, false, false);
        }

        // ---- strictly-lower-triangular intra-block recurrence ----
        #pragma unroll 1
        for (int i = 0; i < 16; ++i) {
            const float ex = (float)exist[cbase + i];
            const float wj =
                (l16 > i) ? bf2f(slab[l16 * NUM_NODES + INPUT_DIM + cbase + i])
                          : 0.0f;
            const int src = i | (lane & 16);     // col i lives in lanes i, i+16
            #pragma unroll
            for (int v = 0; v < 8; ++v) {
                float t = __shfl(acc[v], src, 32);   // ds_bpermute broadcast
                float r = sigm(t) * ex;
                acc[v] += r * wj;
            }
        }

        // ---- finalize: res = sigmoid(acc)*exist -> bf16 signals in LDS ----
        const float exCol = (float)exist[cbase + l16];
        #pragma unroll
        for (int v = 0; v < 8; ++v) {
            float r = sigm(acc[v]) * exCol;
            int m = wave * 16 + half * 8 + v;          // C layout row
            sig[m * NUM_NODES + INPUT_DIM + cbase + l16] = f2bf(r);
        }
        __syncthreads();   // sig published; slab free to be overwritten
    }

    // ---- output: last 64 mid columns, f32 ----
    for (int idx = tid; idx < TILE_M * OUT_DIM; idx += BLOCK_THREADS) {
        int r = idx >> 6, q = idx & 63;
        out[(size_t)(row0 + r) * OUT_DIM + q] =
            bf2f(sig[r * NUM_NODES + INPUT_DIM + (NUM_MID - OUT_DIM) + q]);
    }
}

// ---------------------------------------------------------------------------
extern "C" void kernel_launch(void* const* d_in, const int* in_sizes, int n_in,
                              void* d_out, int out_size, void* d_ws, size_t ws_size,
                              hipStream_t stream) {
    const float* x      = (const float*)d_in[0];
    const float* weight = (const float*)d_in[1];
    const float* bias   = (const float*)d_in[2];
    const int*   conn   = (const int*)d_in[3];
    const int*   exist  = (const int*)d_in[4];
    float* out = (float*)d_out;
    unsigned short* effT = (unsigned short*)d_ws;   // 512*1024*2 = 1 MB

    build_effT<<<(NUM_MID * NUM_NODES + 255) / 256, 256, 0, stream>>>(
        weight, conn, effT);
    recurrent_kernel<<<BATCH / TILE_M, BLOCK_THREADS, 0, stream>>>(
        x, effT, bias, exist, out);
}
